// HGT_67877663146324
// MI455X (gfx1250) — matrix operand order repaired and checked by reading, not compiled, not run on hardware
//
#include <hip/hip_runtime.h>

// ---------------- problem constants ----------------
#define NN   100000   // nodes (6250 * 16, exact WMMA M-tiles)
#define HID  128
#define RR   8
#define LL   2
#define HH   4
#define DD   32
#define EE   625000
#define KQV  (3*HID)  // 384

typedef _Float16 half_t;
typedef __attribute__((ext_vector_type(16))) _Float16 v16h;
typedef __attribute__((ext_vector_type(8)))  float    v8f;

__device__ __forceinline__ float gelu_exact(float x) {
    return 0.5f * x * (1.0f + erff(x * 0.70710678118654752f));
}
// order-preserving float<->uint map for atomicMax-based segment max
__device__ __forceinline__ unsigned enc_ord(float f) {
    unsigned b = __float_as_uint(f);
    return (b & 0x80000000u) ? ~b : (b | 0x80000000u);
}
__device__ __forceinline__ float dec_ord(unsigned u) {
    return (u & 0x80000000u) ? __uint_as_float(u ^ 0x80000000u)
                             : __uint_as_float(~u);
}

// ---- fragment helpers (ISA 7.12.2 layouts, wave32) ----
// A 16x32 f16: lanes0-15 row=lane, halves 0-7 -> K k0..k0+7, 8-15 -> k0+16..23; lanes16-31 K base +8
template <int GELU>
__device__ __forceinline__ v16h load_a_frag(const float* __restrict__ ap) {
    v16h a;
    #pragma unroll
    for (int i = 0; i < 8; ++i) {
        float x0 = ap[i];
        float x1 = ap[i + 16];
        if (GELU) { x0 = gelu_exact(x0); x1 = gelu_exact(x1); }
        a[i]     = (half_t)x0;
        a[i + 8] = (half_t)x1;
    }
    return a;
}
// B 32x16 f16: lane = column; halves 0-7 -> K k0..k0+7, 8-15 -> k0+16..23; lanes16-31 K base +8
__device__ __forceinline__ v16h load_b_frag(const half_t* __restrict__ bp) {
    v16h b;
    #pragma unroll
    for (int i = 0; i < 8; ++i) { b[i] = bp[i]; b[i + 8] = bp[i + 16]; }
    return b;
}

// ---------------- WMMA GEMM: C[MxNc] = A[MxK](f32->f16) * Bt[Nc x K](f16)^T + bias ----------------
// One wave computes a 16 x (16*NT) strip: A fragment loaded once per k-step, NT WMMAs against it.
// gridDim.y = Nc/(16*NT), blockDim=(32,4).
template <int GELU, int NT>
__global__ void wmma_gemm_f16(const float* __restrict__ A, int lda,
                              const half_t* __restrict__ Bt, int ldb,
                              const float* __restrict__ bias,
                              float* __restrict__ C, int ldc,
                              int K, int mtiles) {
    const int lane  = threadIdx.x;          // 0..31
    const int mtile = blockIdx.x * blockDim.y + threadIdx.y;
    if (mtile >= mtiles) return;            // wave-uniform exit (EXEC all-1 inside)
    const int nt0   = blockIdx.y * NT;      // first n-tile of this wave's strip
    const int hi    = lane >> 4;
    const int l15   = lane & 15;
    const int mrow  = mtile * 16 + l15;

    v8f acc[NT];
    #pragma unroll
    for (int t = 0; t < NT; ++t) acc[t] = (v8f){0.f,0.f,0.f,0.f,0.f,0.f,0.f,0.f};

    for (int k0 = 0; k0 < K; k0 += 32) {
        const v16h a = load_a_frag<GELU>(A + (size_t)mrow * lda + k0 + (hi ? 8 : 0));
        #pragma unroll
        for (int t = 0; t < NT; ++t) {
            const int nb = (nt0 + t) * 16 + l15;
            const v16h b = load_b_frag(Bt + (size_t)nb * ldb + k0 + (hi ? 8 : 0));
            acc[t] = __builtin_amdgcn_wmma_f32_16x16x32_f16(
                         false, a, false, b, (short)0, acc[t], false, false);
        }
    }
    // D layout: vgpr r -> row mtile*16 + hi*8 + r, column nb
    #pragma unroll
    for (int t = 0; t < NT; ++t) {
        const int nb = (nt0 + t) * 16 + l15;
        const float bv = bias ? bias[nb] : 0.0f;
        float* cp = C + (size_t)(mtile * 16 + hi * 8) * ldc + nb;
        #pragma unroll
        for (int r = 0; r < 8; ++r) cp[(size_t)r * ldc] = acc[t][r] + bv;
    }
}

// ---------------- batched per-head GEMM (block-diagonal): blockIdx.y = head ----------------
// C[:, hh*32 .. hh*32+31] = A[:, hh*32 .. hh*32+31] @ W[hh]  (K = D = 32, one k-step, 2 n-tiles)
__global__ void wmma_head_gemm(const float* __restrict__ Abase, int lda,
                               const half_t* __restrict__ WT,   // [H][D][D] f16 (pre-transposed)
                               float* __restrict__ C, int ldc, int mtiles) {
    const int lane  = threadIdx.x;
    const int mtile = blockIdx.x * blockDim.y + threadIdx.y;
    if (mtile >= mtiles) return;
    const int hh    = blockIdx.y;           // head
    const int hi    = lane >> 4;
    const int l15   = lane & 15;
    const int mrow  = mtile * 16 + l15;

    const float*  A  = Abase + hh * DD;
    const half_t* Bt = WT + (size_t)hh * DD * DD;

    const v16h a = load_a_frag<0>(A + (size_t)mrow * lda + (hi ? 8 : 0));
    v8f acc0 = (v8f){0.f,0.f,0.f,0.f,0.f,0.f,0.f,0.f};
    v8f acc1 = acc0;
    const v16h b0 = load_b_frag(Bt + (size_t)(l15)      * DD + (hi ? 8 : 0));
    const v16h b1 = load_b_frag(Bt + (size_t)(16 + l15) * DD + (hi ? 8 : 0));
    acc0 = __builtin_amdgcn_wmma_f32_16x16x32_f16(false, a, false, b0, (short)0, acc0, false, false);
    acc1 = __builtin_amdgcn_wmma_f32_16x16x32_f16(false, a, false, b1, (short)0, acc1, false, false);

    float* cp0 = C + (size_t)(mtile * 16 + hi * 8) * ldc + hh * DD + l15;
    #pragma unroll
    for (int r = 0; r < 8; ++r) {
        cp0[(size_t)r * ldc]      = acc0[r];
        cp0[(size_t)r * ldc + 16] = acc1[r];
    }
}

// ---------------- weight prep: f32 -> f16 with transpose ----------------
__global__ void transpose_to_f16(const float* __restrict__ in, half_t* __restrict__ out,
                                 int rows, int cols) {
    int idx = blockIdx.x * blockDim.x + threadIdx.x;
    if (idx >= rows * cols) return;
    int r = idx / cols, c = idx % cols;
    out[(size_t)c * rows + r] = (half_t)in[idx];
}
__global__ void transpose_heads_to_f16(const float* __restrict__ in, half_t* __restrict__ out,
                                       int mats) {
    int idx = blockIdx.x * blockDim.x + threadIdx.x;
    if (idx >= mats * DD * DD) return;
    int m = idx >> 10, o = idx & 1023;
    int d = o >> 5, f = o & 31;
    out[(size_t)m * 1024 + f * DD + d] = (half_t)in[idx];
}

// ---------------- misc elementwise ----------------
__global__ void gather_emb(const float* __restrict__ emb, const int* __restrict__ x,
                           float* __restrict__ h) {
    int idx = blockIdx.x * blockDim.x + threadIdx.x;
    if (idx >= NN * HID) return;
    int n = idx >> 7, c = idx & (HID - 1);
    h[idx] = emb[(size_t)x[n] * HID + c];
}
__global__ void zero_f32(float* __restrict__ p, int n) {
    int idx = blockIdx.x * blockDim.x + threadIdx.x;
    if (idx < n) p[idx] = 0.0f;
}
__global__ void skip_relu(const float* __restrict__ o, const float* __restrict__ skip,
                          int l, float* __restrict__ h) {
    int idx = blockIdx.x * blockDim.x + threadIdx.x;
    if (idx >= NN * HID) return;
    float sg = 1.0f / (1.0f + __expf(-skip[l]));
    float v = sg * o[idx] + (1.0f - sg) * h[idx];
    h[idx] = v > 0.0f ? v : 0.0f;
}

// ---------------- edge passes ----------------
__global__ void edge_alpha(const int* __restrict__ et, const int* __restrict__ src,
                           const int* __restrict__ dst, int rel,
                           const float* __restrict__ qbase,   // kqv + HID, row stride KQV
                           const float* __restrict__ kp,      // [N][HID]
                           const float* __restrict__ prel,    // [R][H]
                           float* __restrict__ alpha, unsigned* __restrict__ amax) {
    int e = blockIdx.x * blockDim.x + threadIdx.x;
    if (e >= EE || et[e] != rel) return;
    int s = src[e], d = dst[e];
    const float4* qp  = (const float4*)(qbase + (size_t)d * KQV);
    const float4* kpp = (const float4*)(kp    + (size_t)s * HID);
    const float scale = 0.17677669529663687f;  // 1/sqrt(D)
    #pragma unroll
    for (int h = 0; h < HH; ++h) {
        float acc = 0.0f;
        #pragma unroll
        for (int i = 0; i < 8; ++i) {
            float4 q4 = qp[h * 8 + i];
            float4 k4 = kpp[h * 8 + i];
            acc += q4.x * k4.x + q4.y * k4.y + q4.z * k4.z + q4.w * k4.w;
        }
        float a = acc * prel[rel * HH + h] * scale;
        alpha[(size_t)e * HH + h] = a;
        atomicMax(&amax[(size_t)d * HH + h], enc_ord(a));
    }
}
__global__ void edge_softmax(const int* __restrict__ dst, float* __restrict__ alpha,
                             const unsigned* __restrict__ amax, float* __restrict__ denom) {
    int e = blockIdx.x * blockDim.x + threadIdx.x;
    if (e >= EE) return;
    int d = dst[e];
    #pragma unroll
    for (int h = 0; h < HH; ++h) {
        float m  = dec_ord(amax[(size_t)d * HH + h]);
        float ee = __expf(alpha[(size_t)e * HH + h] - m);
        alpha[(size_t)e * HH + h] = ee;
        atomicAdd(&denom[(size_t)d * HH + h], ee);
    }
}
__global__ void edge_aggr(const int* __restrict__ et, const int* __restrict__ src,
                          const int* __restrict__ dst, int rel,
                          const float* __restrict__ vp, const float* __restrict__ alpha,
                          const float* __restrict__ denom, float* __restrict__ aggr) {
    int e = blockIdx.x * blockDim.x + threadIdx.x;
    if (e >= EE || et[e] != rel) return;
    int s = src[e], d = dst[e];
    const float4* vpp = (const float4*)(vp + (size_t)s * HID);
    #pragma unroll
    for (int h = 0; h < HH; ++h) {
        float w = alpha[(size_t)e * HH + h] / denom[(size_t)d * HH + h];
        #pragma unroll
        for (int i = 0; i < 8; ++i) {
            float4 v4 = vpp[h * 8 + i];
            float* ab = aggr + (size_t)d * HID + h * DD + i * 4;
            atomicAdd(ab + 0, v4.x * w);
            atomicAdd(ab + 1, v4.y * w);
            atomicAdd(ab + 2, v4.z * w);
            atomicAdd(ab + 3, v4.w * w);
        }
    }
}

// ---------------- host orchestration ----------------
extern "C" void kernel_launch(void* const* d_in, const int* in_sizes, int n_in,
                              void* d_out, int out_size, void* d_ws, size_t ws_size,
                              hipStream_t stream) {
    const float* emb    = (const float*)d_in[0];
    const float* W_kqv  = (const float*)d_in[1];
    const float* b_kqv  = (const float*)d_in[2];
    const float* Wk     = (const float*)d_in[3];
    const float* Wv     = (const float*)d_in[4];
    const float* p_rel  = (const float*)d_in[5];
    const float* W_out  = (const float*)d_in[6];
    const float* b_out  = (const float*)d_in[7];
    const float* skip   = (const float*)d_in[8];
    const int*   x      = (const int*)d_in[9];
    const int*   src    = (const int*)d_in[10];        // edge_index[0]
    const int*   dst    = ((const int*)d_in[10]) + EE; // edge_index[1]
    const int*   etype  = (const int*)d_in[11];
    float*       h      = (float*)d_out;               // [N][HID], persists across layers

    // ---- carve workspace (256B aligned) ----
    uintptr_t base = (uintptr_t)d_ws;
    auto alloc = [&](size_t bytes) -> void* {
        base = (base + 255) & ~(uintptr_t)255;
        void* p = (void*)base;
        base += bytes;
        return p;
    };
    float*    kqv   = (float*)   alloc((size_t)NN * KQV * 4);   // reused as obuf
    float*    tbuf  = (float*)   alloc((size_t)NN * HID * 4);   // kp_r / vp_r
    float*    alpha = (float*)   alloc((size_t)EE * HH * 4);
    unsigned* amax  = (unsigned*)alloc((size_t)NN * HH * 4);    // amax,denom,aggr contiguous
    float*    denom = (float*)   alloc((size_t)NN * HH * 4);
    float*    aggr  = (float*)   alloc((size_t)NN * HID * 4);
    half_t*   wkqvT = (half_t*)  alloc((size_t)KQV * HID * 2);
    half_t*   wkT   = (half_t*)  alloc((size_t)RR * HH * DD * DD * 2);
    half_t*   wvT   = (half_t*)  alloc((size_t)RR * HH * DD * DD * 2);
    half_t*   woutT = (half_t*)  alloc((size_t)HID * HID * 2);
    float*    obuf  = kqv;  // alias: kqv dead by the time out-proj runs
    (void)in_sizes; (void)n_in; (void)out_size; (void)ws_size;

    const int MT = NN / 16;                 // 6250 exact M tiles
    dim3 gblk(32, 4);                       // 4 waves per block
    const int gx = (MT + 3) / 4;
    const int eb = (EE + 255) / 256;

    gather_emb<<<(NN * HID + 255) / 256, 256, 0, stream>>>(emb, x, h);

    for (int l = 0; l < LL; ++l) {
        // weight prep (f32 -> f16, transposed so B fragments are contiguous)
        transpose_to_f16<<<(HID * KQV + 255) / 256, 256, 0, stream>>>(
            W_kqv + (size_t)l * HID * KQV, wkqvT, HID, KQV);
        transpose_heads_to_f16<<<(RR * HH * DD * DD + 255) / 256, 256, 0, stream>>>(
            Wk + (size_t)l * RR * HH * DD * DD, wkT, RR * HH);
        transpose_heads_to_f16<<<(RR * HH * DD * DD + 255) / 256, 256, 0, stream>>>(
            Wv + (size_t)l * RR * HH * DD * DD, wvT, RR * HH);
        transpose_to_f16<<<(HID * HID + 255) / 256, 256, 0, stream>>>(
            W_out + (size_t)l * HID * HID, woutT, HID, HID);

        // kqv = h @ W_kqv + b  (WMMA, 16x64 strips: grid.y = 384/64 = 6)
        wmma_gemm_f16<0, 4><<<dim3(gx, KQV / 64), gblk, 0, stream>>>(
            h, HID, wkqvT, HID, b_kqv + (size_t)l * KQV, kqv, KQV, HID, MT);

        // zero amax | denom | aggr (contiguous span)
        {
            int zn = NN * HH + NN * HH + NN * HID;
            zero_f32<<<(zn + 255) / 256, 256, 0, stream>>>((float*)amax, zn);
        }

        // pass A: per relation, kp_r = k @ Wk[r] (batched per-head WMMA), then logits + seg-max
        for (int r = 0; r < RR; ++r) {
            wmma_head_gemm<<<dim3(gx, HH), gblk, 0, stream>>>(
                kqv, KQV, wkT + (size_t)r * HH * DD * DD, tbuf, HID, MT);
            edge_alpha<<<eb, 256, 0, stream>>>(etype, src, dst, r, kqv + HID, tbuf,
                                               p_rel + (size_t)l * RR * HH, alpha, amax);
        }

        // segment softmax numerators + denominators
        edge_softmax<<<eb, 256, 0, stream>>>(dst, alpha, amax, denom);

        // pass B: per relation, vp_r = v @ Wv[r] (batched per-head WMMA), then weighted scatter
        for (int r = 0; r < RR; ++r) {
            wmma_head_gemm<<<dim3(gx, HH), gblk, 0, stream>>>(
                kqv + 2 * HID, KQV, wvT + (size_t)r * HH * DD * DD, tbuf, HID, MT);
            edge_aggr<<<eb, 256, 0, stream>>>(etype, src, dst, r, tbuf, alpha, denom, aggr);
        }

        // o = gelu(aggr) @ W_out + b_out  (gelu fused into A-fragment load; grid.y = 128/64 = 2)
        wmma_gemm_f16<1, 4><<<dim3(gx, HID / 64), gblk, 0, stream>>>(
            aggr, HID, woutT, HID, b_out + (size_t)l * HID, obuf, HID, HID, MT);

        // h = relu(sg*o + (1-sg)*h)
        skip_relu<<<(NN * HID + 255) / 256, 256, 0, stream>>>(obuf, skip, l, h);
    }
}
